// Decoder_89404039233716
// MI455X (gfx1250) — matrix operand order repaired and checked by reading, not compiled, not run on hardware
//
#include <hip/hip_runtime.h>

typedef __bf16 bf16;
typedef __attribute__((ext_vector_type(16))) __bf16 v16bf;
typedef __attribute__((ext_vector_type(8)))  __bf16 v8bf;
typedef __attribute__((ext_vector_type(8)))  float  v8f;

// ---------------- problem constants ----------------
#define BB 32
#define TT 64
#define SS 64
#define VV 32000
#define EE 512
#define UU 1024
#define KZ 2560      // E + U + U  : [emb ; a_prev ; h_prev]
#define NZ 4096      // 4*U gates
#define START_ID 1

// =====================================================================
// WMMA fragment loads (ISA 7.12.2 layouts), bf16, K step = 32.
// A row-major [M][K]; B supplied as BT = B^T row-major [N][K].
// A-frag: lane m=lane&15, grp=lane>>4 -> K {grp*8..+7} and {grp*8+16..+7}
// B-frag: lane n=lane&15, grp -> K {grp*8..+7} and {grp*8+16..+7} in BT row.
// =====================================================================
__device__ __forceinline__ v16bf load_afrag(const bf16* p) {
  v8bf lo = *(const v8bf*)(p);
  v8bf hi = *(const v8bf*)(p + 16);
  v16bf r;
#pragma unroll
  for (int i = 0; i < 8; ++i) { r[i] = lo[i]; r[i + 8] = hi[i]; }
  return r;
}
__device__ __forceinline__ v16bf load_bfrag(const bf16* p) {
  v8bf lo = *(const v8bf*)(p);
  v8bf hi = *(const v8bf*)(p + 16);
  v16bf r;
#pragma unroll
  for (int i = 0; i < 8; ++i) { r[i] = lo[i]; r[i + 8] = hi[i]; }
  return r;
}

// Register-blocked GEMM core: each wave computes (MT*16) x (NT*16).
template <int MT, int NT>
__device__ __forceinline__ void gemm_core(const bf16* __restrict__ A, int lda,
                                          const bf16* __restrict__ BT, int ldbt,
                                          int K, int mb, int nb,
                                          v8f acc[MT][NT]) {
  const unsigned lane = threadIdx.x & 31u;
  const unsigned m    = lane & 15u;
  const unsigned grp  = lane >> 4;
  const bf16* pa[MT];
  const bf16* pb[NT];
#pragma unroll
  for (int mi = 0; mi < MT; ++mi)
    pa[mi] = A + ((size_t)(mb * MT + mi) * 16 + m) * lda + grp * 8;
#pragma unroll
  for (int ni = 0; ni < NT; ++ni)
    pb[ni] = BT + ((size_t)(nb * NT + ni) * 16 + m) * ldbt + grp * 8;

  for (int k = 0; k < K; k += 32) {
    v16bf av[MT];
#pragma unroll
    for (int mi = 0; mi < MT; ++mi) av[mi] = load_afrag(pa[mi] + k);
#pragma unroll
    for (int ni = 0; ni < NT; ++ni) {
      v16bf bv = load_bfrag(pb[ni] + k);   // consumed immediately (low pressure)
#pragma unroll
      for (int mi = 0; mi < MT; ++mi)
        acc[mi][ni] = __builtin_amdgcn_wmma_f32_16x16x32_bf16(
            false, av[mi], false, bv, (short)0, acc[mi][ni], false, false);
    }
  }
}

// f32 output + optional bias
template <int MT, int NT>
__global__ __launch_bounds__(256, 1)
void k_gemm_f32_t(const bf16* __restrict__ A, int lda,
                  const bf16* __restrict__ BT, int ldbt,
                  const float* __restrict__ bias,
                  float* __restrict__ D, int ldd,
                  int Mblk, int Nblk, int K) {
  int wave = (int)((blockIdx.x * blockDim.x + threadIdx.x) >> 5);
  if (wave >= Mblk * Nblk) return;           // wave-uniform guard
  int mb = wave / Nblk, nb = wave % Nblk;
  v8f acc[MT][NT] = {};
  gemm_core<MT, NT>(A, lda, BT, ldbt, K, mb, nb, acc);
  unsigned lane = threadIdx.x & 31u;
  unsigned n = lane & 15u, grp = lane >> 4;
#pragma unroll
  for (int ni = 0; ni < NT; ++ni) {
    int col = (nb * NT + ni) * 16 + n;
    float bv = bias ? bias[col] : 0.f;
#pragma unroll
    for (int mi = 0; mi < MT; ++mi) {
#pragma unroll
      for (int r = 0; r < 8; ++r) {
        int row = (mb * MT + mi) * 16 + grp * 8 + r;
        D[(size_t)row * ldd + col] = acc[mi][ni][r] + bv;
      }
    }
  }
}

// bf16 output with custom row stride (for attn_seq[b][t][:])
template <int MT, int NT>
__global__ __launch_bounds__(256, 1)
void k_gemm_bf16out_t(const bf16* __restrict__ A, int lda,
                      const bf16* __restrict__ BT, int ldbt,
                      bf16* __restrict__ D, long rowStride,
                      int Mblk, int Nblk, int K) {
  int wave = (int)((blockIdx.x * blockDim.x + threadIdx.x) >> 5);
  if (wave >= Mblk * Nblk) return;
  int mb = wave / Nblk, nb = wave % Nblk;
  v8f acc[MT][NT] = {};
  gemm_core<MT, NT>(A, lda, BT, ldbt, K, mb, nb, acc);
  unsigned lane = threadIdx.x & 31u;
  unsigned n = lane & 15u, grp = lane >> 4;
#pragma unroll
  for (int ni = 0; ni < NT; ++ni) {
    int col = (nb * NT + ni) * 16 + n;
#pragma unroll
    for (int mi = 0; mi < MT; ++mi) {
#pragma unroll
      for (int r = 0; r < 8; ++r) {
        int row = (mb * MT + mi) * 16 + grp * 8 + r;
        D[(size_t)row * rowStride + col] = (bf16)acc[mi][ni][r];
      }
    }
  }
}

// ------------------- weight prep kernels -------------------
// src fp32 [K][N] row-major -> dst bf16 [N][K]
__global__ void k_transpose_bf16(const float* __restrict__ src,
                                 bf16* __restrict__ dst, int K, int N) {
  long idx = (long)blockIdx.x * blockDim.x + threadIdx.x;
  long total = (long)K * N;
  if (idx >= total) return;
  int n = (int)(idx % N);
  long k = idx / N;
  dst[(long)n * K + k] = (bf16)src[idx];        // coalesced read
}

// WzT[n][k] = k<1536 ? Wx[k][n] : Wh[k-1536][n]
__global__ void k_build_wzT(const float* __restrict__ Wx,
                            const float* __restrict__ Wh,
                            bf16* __restrict__ dst) {
  long idx = (long)blockIdx.x * blockDim.x + threadIdx.x;
  if (idx >= (long)NZ * KZ) return;
  int k = (int)(idx % KZ);
  int n = (int)(idx / KZ);
  float v = (k < (EE + UU)) ? Wx[(long)k * NZ + n]
                            : Wh[(long)(k - (EE + UU)) * NZ + n];
  dst[idx] = (bf16)v;
}

__global__ void k_f32_to_bf16(const float* __restrict__ src,
                              bf16* __restrict__ dst, long total) {
  long idx = (long)blockIdx.x * blockDim.x + threadIdx.x;
  if (idx < total) dst[idx] = (bf16)src[idx];
}

// embedding gather with <start> prepend: out[b][t][e]
__global__ void k_embed(const int* __restrict__ y,
                        const float* __restrict__ emb,
                        bf16* __restrict__ out) {
  long idx = (long)blockIdx.x * blockDim.x + threadIdx.x;
  if (idx >= (long)BB * TT * EE) return;
  int e = (int)(idx % EE);
  int t = (int)((idx / EE) % TT);
  int b = (int)(idx / ((long)EE * TT));
  int tok = (t == 0) ? START_ID : y[b * TT + t - 1];
  out[idx] = (bf16)emb[(long)tok * EE + e];
}

// state init: c = c0 ; hc[:, :U] = bf16(h0)
__global__ void k_init_state(const float* __restrict__ h0,
                             const float* __restrict__ c0,
                             float* __restrict__ c, bf16* __restrict__ hc) {
  int idx = blockIdx.x * blockDim.x + threadIdx.x;
  if (idx >= BB * UU) return;
  int b = idx / UU, u = idx % UU;
  c[idx] = c0[idx];
  hc[b * 2 * UU + u] = (bf16)h0[idx];
}

// xh[b] = [emb[b][t] ; a_prev[b] ; h_prev[b]]  (all bf16)
__global__ void k_build_xh(const bf16* __restrict__ embbf,
                           const bf16* __restrict__ aseq,
                           const bf16* __restrict__ hc,
                           bf16* __restrict__ xh, int t) {
  int idx = blockIdx.x * blockDim.x + threadIdx.x;
  if (idx >= BB * KZ) return;
  int b = idx / KZ, j = idx % KZ;
  bf16 v;
  if (j < EE)
    v = embbf[((long)b * TT + t) * EE + j];
  else if (j < EE + UU)
    v = (t > 0) ? aseq[((long)b * TT + (t - 1)) * UU + (j - EE)] : (bf16)0.f;
  else
    v = hc[b * 2 * UU + (j - EE - UU)];
  xh[idx] = v;
}

// LSTM gates (TF order i,j,f,o; forget_bias=1): updates c, writes h->hc[:, :U]
__global__ void k_lstm_gate(const float* __restrict__ z,
                            float* __restrict__ c, bf16* __restrict__ hc) {
  int idx = blockIdx.x * blockDim.x + threadIdx.x;
  if (idx >= BB * UU) return;
  int b = idx >> 10, u = idx & (UU - 1);
  const float* zb = z + (long)b * NZ;
  float gi = zb[u], gj = zb[UU + u], gf = zb[2 * UU + u], go = zb[3 * UU + u];
  float si = 1.f / (1.f + __expf(-gi));
  float sf = 1.f / (1.f + __expf(-(gf + 1.f)));
  float so = 1.f / (1.f + __expf(-go));
  float cn = sf * c[idx] + si * tanhf(gj);
  c[idx] = cn;
  hc[b * 2 * UU + u] = (bf16)(so * tanhf(cn));
}

// score[b][s] = sum_u v[u] * tanh(keys[b][s][u] + q[b][u])
__global__ void k_attn_score(const float* __restrict__ keys,
                             const float* __restrict__ q,
                             const float* __restrict__ vatt,
                             float* __restrict__ scores) {
  int bs = blockIdx.x;               // b*S + s
  int b = bs >> 6;
  __shared__ float red[256];
  const float* kk = keys + (long)bs * UU;
  const float* qq = q + (long)b * UU;
  float p = 0.f;
  for (int u = threadIdx.x; u < UU; u += 256)
    p += vatt[u] * tanhf(kk[u] + qq[u]);
  red[threadIdx.x] = p;
  __syncthreads();
  for (int s2 = 128; s2 > 0; s2 >>= 1) {
    if ((int)threadIdx.x < s2) red[threadIdx.x] += red[threadIdx.x + s2];
    __syncthreads();
  }
  if (threadIdx.x == 0) scores[bs] = red[0];
}

// softmax over S, ctx = al @ enc ; writes bf16 ctx into hc[:, U:2U]
__global__ void k_attn_ctx(const float* __restrict__ scores,
                           const float* __restrict__ enc,
                           bf16* __restrict__ hc) {
  int b = blockIdx.x;
  __shared__ float al[SS];
  if (threadIdx.x < SS) al[threadIdx.x] = scores[b * SS + threadIdx.x];
  __syncthreads();
  if (threadIdx.x == 0) {
    float m = al[0];
    for (int s = 1; s < SS; ++s) m = fmaxf(m, al[s]);
    float sum = 0.f;
    for (int s = 0; s < SS; ++s) { al[s] = __expf(al[s] - m); sum += al[s]; }
    float inv = 1.f / sum;
    for (int s = 0; s < SS; ++s) al[s] *= inv;
  }
  __syncthreads();
  const float* eb = enc + (long)b * SS * UU;
  for (int e = threadIdx.x; e < UU; e += 256) {
    float acc = 0.f;
    for (int s = 0; s < SS; ++s) acc += al[s] * eb[(long)s * UU + e];
    hc[b * 2 * UU + UU + e] = (bf16)acc;
  }
}

// =====================================================================
extern "C" void kernel_launch(void* const* d_in, const int* in_sizes, int n_in,
                              void* d_out, int out_size, void* d_ws, size_t ws_size,
                              hipStream_t stream) {
  const int*   y     = (const int*)  d_in[0];
  const float* h0    = (const float*)d_in[1];
  const float* c0    = (const float*)d_in[2];
  const float* enc   = (const float*)d_in[3];
  const float* embed = (const float*)d_in[4];
  const float* Wx    = (const float*)d_in[5];
  const float* Wh    = (const float*)d_in[6];
  const float* bvec  = (const float*)d_in[7];
  const float* Wk    = (const float*)d_in[8];
  const float* Wq    = (const float*)d_in[9];
  const float* vatt  = (const float*)d_in[10];
  const float* Wa    = (const float*)d_in[11];
  const float* Wout  = (const float*)d_in[12];
  const float* bout  = (const float*)d_in[13];
  float* out = (float*)d_out;

  // ---- workspace carve-up (256B aligned) ----
  char* ws = (char*)d_ws;
  size_t off = 0;
  auto carve = [&](size_t bytes) -> char* {
    char* p = ws + off;
    off = (off + bytes + 255) & ~(size_t)255;
    return p;
  };
  bf16*  wzT   = (bf16*) carve((size_t)NZ * KZ * 2);       // [4096][2560]
  bf16*  wkT   = (bf16*) carve((size_t)UU * UU * 2);       // [1024][1024]
  bf16*  wqT   = (bf16*) carve((size_t)UU * UU * 2);
  bf16*  waT   = (bf16*) carve((size_t)UU * 2 * UU * 2);   // [1024][2048]
  bf16*  woT   = (bf16*) carve((size_t)VV * UU * 2);       // [32000][1024]
  bf16*  encbf = (bf16*) carve((size_t)BB * SS * UU * 2);
  float* keys  = (float*)carve((size_t)BB * SS * UU * 4);
  bf16*  embbf = (bf16*) carve((size_t)BB * TT * EE * 2);
  bf16*  aseq  = (bf16*) carve((size_t)BB * TT * UU * 2);  // attention outputs
  bf16*  xh    = (bf16*) carve((size_t)BB * KZ * 2);
  float* zbuf  = (float*)carve((size_t)BB * NZ * 4);
  float* cbuf  = (float*)carve((size_t)BB * UU * 4);
  float* qbuf  = (float*)carve((size_t)BB * UU * 4);
  bf16*  hc    = (bf16*) carve((size_t)BB * 2 * UU * 2);   // [h ; ctx] bf16
  float* scr   = (float*)carve((size_t)BB * SS * 4);

  auto g1 = [](long n) { return (unsigned)((n + 255) / 256); };     // 256-thr grids
  auto gt = [](long waves) { return (unsigned)((waves + 7) / 8); }; // 8 waves/blk

  // ---- one-time weight conversion / transposition ----
  k_build_wzT     <<<g1((long)NZ * KZ), 256, 0, stream>>>(Wx, Wh, wzT);
  k_transpose_bf16<<<g1((long)UU * UU), 256, 0, stream>>>(Wk, wkT, UU, UU);
  k_transpose_bf16<<<g1((long)UU * UU), 256, 0, stream>>>(Wq, wqT, UU, UU);
  k_transpose_bf16<<<g1((long)2 * UU * UU), 256, 0, stream>>>(Wa, waT, 2 * UU, UU);
  k_transpose_bf16<<<g1((long)UU * VV), 256, 0, stream>>>(Wout, woT, UU, VV);
  k_f32_to_bf16   <<<g1((long)BB * SS * UU), 256, 0, stream>>>(enc, encbf, (long)BB * SS * UU);
  k_embed         <<<g1((long)BB * TT * EE), 256, 0, stream>>>(y, embed, embbf);
  k_init_state    <<<g1(BB * UU), 256, 0, stream>>>(h0, c0, cbuf, hc);

  // ---- keys = enc @ Wk : [2048][1024], K=1024 ; 4x2 blocking ----
  {
    int Mblk = (BB * SS) / 64, Nblk = UU / 32;   // 32 x 32 = 1024 waves
    k_gemm_f32_t<4, 2><<<gt((long)Mblk * Nblk), 256, 0, stream>>>(
        encbf, UU, wkT, UU, nullptr, keys, UU, Mblk, Nblk, UU);
  }

  // ---- recurrent scan over T ----
  for (int t = 0; t < TT; ++t) {
    k_build_xh<<<g1(BB * KZ), 256, 0, stream>>>(embbf, aseq, hc, xh, t);
    // z = xh @ Wz + b : [32][4096], K=2560 ; 2x4 blocking (Wz read once)
    {
      int Mblk = BB / 32, Nblk = NZ / 64;        // 1 x 64 waves
      k_gemm_f32_t<2, 4><<<gt((long)Mblk * Nblk), 256, 0, stream>>>(
          xh, KZ, wzT, KZ, bvec, zbuf, NZ, Mblk, Nblk, KZ);
    }
    k_lstm_gate<<<g1(BB * UU), 256, 0, stream>>>(zbuf, cbuf, hc);
    // q = h @ Wq : [32][1024], K=1024  (A = hc[:, :U], lda = 2U)
    {
      int Mblk = BB / 32, Nblk = UU / 64;        // 1 x 16 waves
      k_gemm_f32_t<2, 4><<<gt((long)Mblk * Nblk), 256, 0, stream>>>(
          hc, 2 * UU, wqT, UU, nullptr, qbuf, UU, Mblk, Nblk, UU);
    }
    k_attn_score<<<BB * SS, 256, 0, stream>>>(keys, qbuf, vatt, scr);
    k_attn_ctx  <<<BB, 256, 0, stream>>>(scr, enc, hc);
    // a = [h;ctx] @ Wa : [32][1024], K=2048 -> aseq[:, t, :]
    {
      int Mblk = BB / 32, Nblk = UU / 64;
      k_gemm_bf16out_t<2, 4><<<gt((long)Mblk * Nblk), 256, 0, stream>>>(
          hc, 2 * UU, waT, 2 * UU, aseq + (size_t)t * UU, (long)TT * UU,
          Mblk, Nblk, 2 * UU);
    }
  }

  // ---- logits = attn_seq @ Wout + bout : [2048][32000], K=1024 ; 4x2 ----
  {
    int Mblk = (BB * TT) / 64, Nblk = VV / 32;   // 32 x 1000 = 32000 waves
    k_gemm_f32_t<4, 2><<<gt((long)Mblk * Nblk), 256, 0, stream>>>(
        aseq, UU, woT, UU, bout, out, VV, Mblk, Nblk, UU);
  }
}
